// DualInputGRUCell_63814624084652
// MI455X (gfx1250) — matrix operand order repaired and checked by reading, not compiled, not run on hardware
//
#include <hip/hip_runtime.h>

typedef __attribute__((ext_vector_type(16))) __bf16 v16bf;
typedef __attribute__((ext_vector_type(8)))  float  v8f;
typedef __attribute__((ext_vector_type(4)))  float  v4f;
typedef __attribute__((ext_vector_type(8)))  unsigned short v8us;
typedef __attribute__((ext_vector_type(16))) unsigned short v16us;

#define BATCH 131072
#define DHID  128
#define MATSZ (DHID * DHID)  // 16384 elements per weight matrix

__device__ __forceinline__ v8f wmma_bf16(v16bf a, v16bf b, v8f c) {
  // D = A(16x32 bf16) * B(32x16 bf16) + C(16x16 f32)
  return __builtin_amdgcn_wmma_f32_16x16x32_bf16(false, a, false, b, (short)0, c,
                                                 false, false);
}

__device__ __forceinline__ unsigned short f2bf(float f) {
  return __builtin_bit_cast(unsigned short, (__bf16)f);
}

// A-fragment from global fp32 row-major [., 128].
// ISA 16-bit A layout (16x32): lanes 0-15 row M=lane, elems 0..7 = K 0..7,
// elems 8..15 = K 16..23; lanes 16-31 same rows, K base +8.
__device__ __forceinline__ v16bf load_a_g(const float* __restrict__ X, int row0,
                                          int lane, int kt) {
  const int m   = row0 + (lane & 15);
  const int klo = (lane & 16) ? 8 : 0;
  const float* p = X + (size_t)m * DHID + kt * 32 + klo;
  v4f f0 = *(const v4f*)(p);
  v4f f1 = *(const v4f*)(p + 4);
  v4f f2 = *(const v4f*)(p + 16);
  v4f f3 = *(const v4f*)(p + 20);
  v16us t;
#pragma unroll
  for (int i = 0; i < 4; ++i) {
    t[i]      = f2bf(f0[i]);
    t[4 + i]  = f2bf(f1[i]);
    t[8 + i]  = f2bf(f2[i]);
    t[12 + i] = f2bf(f3[i]);
  }
  return __builtin_bit_cast(v16bf, t);
}

// B-fragment from bf16 weights stored transposed as [N=128][K=128].
// ISA B layout (32x16): lane n = lane&15; lanes 0-15 hold K 0..7, 16..23,
// lanes 16-31 hold K 8..15, 24..31 (matches A fragment order).
__device__ __forceinline__ v16bf load_b(const unsigned short* __restrict__ Wt,
                                        int lane, int jt, int kt) {
  const int n  = jt * 16 + (lane & 15);
  const int kb = kt * 32 + ((lane & 16) ? 8 : 0);
  const unsigned short* p = Wt + n * DHID + kb;
  v8us lo = *(const v8us*)(p);
  v8us hi = *(const v8us*)(p + 16);
  v16us t;
#pragma unroll
  for (int i = 0; i < 8; ++i) { t[i] = lo[i]; t[8 + i] = hi[i]; }
  return __builtin_bit_cast(v16bf, t);
}

// A-fragment from per-wave LDS bf16 tile [16][128]
__device__ __forceinline__ v16bf load_a_lds(const unsigned short* S, int lane,
                                            int kt) {
  const int m   = lane & 15;
  const int klo = (lane & 16) ? 8 : 0;
  const unsigned short* p = S + m * DHID + kt * 32 + klo;
  v8us a0 = *(const v8us*)(p);
  v8us a1 = *(const v8us*)(p + 16);
  v16us t;
#pragma unroll
  for (int i = 0; i < 8; ++i) { t[i] = a0[i]; t[8 + i] = a1[i]; }
  return __builtin_bit_cast(v16bf, t);
}

// elementwise bf16 product of two A-layout fragments (layouts match per lane)
__device__ __forceinline__ v16bf bf_mul(v16bf a, v16bf b) {
  v16bf c;
#pragma unroll
  for (int i = 0; i < 16; ++i) c[i] = a[i] * b[i];
  return c;
}

__device__ __forceinline__ float sigm(float x) { return 1.0f / (1.0f + __expf(-x)); }

__device__ __forceinline__ float tanh_fast(float x) {
  float xc = fminf(fmaxf(x, -15.0f), 15.0f);
  float e  = __expf(2.0f * xc);
  return (e - 1.0f) / (e + 1.0f);
}

// ---- prepass: W [K=128][N=128] fp32 -> Wt [N=128][K=128] bf16 ----
__global__ void transpose_to_bf16(const float* __restrict__ W,
                                  unsigned short* __restrict__ Wt) {
  int idx = blockIdx.x * blockDim.x + threadIdx.x;  // 0..16383
  int n = idx >> 7;
  int k = idx & 127;
  Wt[idx] = f2bf(W[k * DHID + n]);
}

// ---- fused dual-input GRU cell ----
__global__ __launch_bounds__(128) void gru_fused(
    const float* __restrict__ X, const float* __restrict__ S,
    const float* __restrict__ H, const unsigned short* __restrict__ Wt,
    const float* __restrict__ b_ir, const float* __restrict__ b_iz,
    const float* __restrict__ b_in, const float* __restrict__ b_sr,
    const float* __restrict__ b_sz, const float* __restrict__ b_sn,
    const float* __restrict__ b_hr, const float* __restrict__ b_hz,
    const float* __restrict__ b_hn, float* __restrict__ out) {
  // per-wave LDS: z (16x128 f32) + n_pre (16x128 f32) + r (16x128 bf16)
  __shared__ __align__(16) unsigned char smem_raw[4 * (8192 + 8192 + 4096)];

  const int lane = threadIdx.x & 31;
  const int w    = threadIdx.x >> 5;
  const int row0 = (blockIdx.x * 4 + w) * 16;

  float*          zs  = (float*)(smem_raw + w * 20480);
  float*          nps = (float*)(smem_raw + w * 20480 + 8192);
  unsigned short* rbs = (unsigned short*)(smem_raw + w * 20480 + 16384);

  const int mlo = lane & 15;
  const int hi8 = (lane & 16) ? 8 : 0;

  // weight matrices in Wt order: ir iz in sr sz sn hr hz hn
  const unsigned short* Wir = Wt + 0 * MATSZ;
  const unsigned short* Wiz = Wt + 1 * MATSZ;
  const unsigned short* Win = Wt + 2 * MATSZ;
  const unsigned short* Wsr = Wt + 3 * MATSZ;
  const unsigned short* Wsz = Wt + 4 * MATSZ;
  const unsigned short* Wsn = Wt + 5 * MATSZ;
  const unsigned short* Whr = Wt + 6 * MATSZ;
  const unsigned short* Whz = Wt + 7 * MATSZ;
  const unsigned short* Whn = Wt + 8 * MATSZ;

  // ---- phase 1: A fragments for the three activations ----
  v16bf ain[4], asq[4], ahd[4];
#pragma unroll
  for (int kt = 0; kt < 4; ++kt) {
    ain[kt] = load_a_g(X, row0, lane, kt);
    asq[kt] = load_a_g(S, row0, lane, kt);
    ahd[kt] = load_a_g(H, row0, lane, kt);
  }

  // ---- phase 2: the 8 shared-gate GEMMs + r/z combine ----
  for (int jt = 0; jt < 8; ++jt) {
    // zero accumulators -> first WMMA uses inline SRC2 = 0
    v8f hr = {}, hz = {}, ir = {}, iz = {}, inn = {}, sr = {}, sz = {}, sn = {};
#pragma unroll
    for (int kt = 0; kt < 4; ++kt) {
      hr  = wmma_bf16(ahd[kt], load_b(Whr, lane, jt, kt), hr);
      hz  = wmma_bf16(ahd[kt], load_b(Whz, lane, jt, kt), hz);
      ir  = wmma_bf16(ain[kt], load_b(Wir, lane, jt, kt), ir);
      iz  = wmma_bf16(ain[kt], load_b(Wiz, lane, jt, kt), iz);
      inn = wmma_bf16(ain[kt], load_b(Win, lane, jt, kt), inn);
      sr  = wmma_bf16(asq[kt], load_b(Wsr, lane, jt, kt), sr);
      sz  = wmma_bf16(asq[kt], load_b(Wsz, lane, jt, kt), sz);
      sn  = wmma_bf16(asq[kt], load_b(Wsn, lane, jt, kt), sn);
    }
    const int col = jt * 16 + mlo;
    // fold biases into the combine (one scalar per lane per gate-pair)
    const float bhr   = b_hr[col];
    const float bhz   = b_hz[col];
    const float birhr = b_ir[col] + bhr;
    const float bsrhr = b_sr[col] + bhr;
    const float bizhz = b_iz[col] + bhz;
    const float bszhz = b_sz[col] + bhz;
    const float binsn = b_in[col] + b_sn[col];
#pragma unroll
    for (int v = 0; v < 8; ++v) {
      const int m = v + hi8;
      float r = 0.5f * (sigm(ir[v] + hr[v] + birhr) + sigm(sr[v] + hr[v] + bsrhr));
      float z = 0.5f * (sigm(iz[v] + hz[v] + bizhz) + sigm(sz[v] + hz[v] + bszhz));
      zs[m * DHID + col]  = z;
      nps[m * DHID + col] = inn[v] + sn[v] + binsn;
      rbs[m * DHID + col] = f2bf(r);
    }
  }

  // ensure all per-wave LDS writes landed before re-reading as A fragments
  asm volatile("s_wait_dscnt 0" ::: "memory");

  // ---- phase 3: hn GEMM on (r*hidden) + final blend ----
  // r comes back from LDS in A-layout; multiply elementwise with the hidden
  // A-fragments already in registers (identical per-lane layout) -> r*h
  v16bf arh[4];
#pragma unroll
  for (int kt = 0; kt < 4; ++kt)
    arh[kt] = bf_mul(load_a_lds(rbs, lane, kt), ahd[kt]);

  for (int jt = 0; jt < 8; ++jt) {
    v8f hn = {};
#pragma unroll
    for (int kt = 0; kt < 4; ++kt)
      hn = wmma_bf16(arh[kt], load_b(Whn, lane, jt, kt), hn);
    const int col = jt * 16 + mlo;
    const float bhn = b_hn[col];
#pragma unroll
    for (int v = 0; v < 8; ++v) {
      const int m  = v + hi8;
      float z  = zs[m * DHID + col];
      float np = nps[m * DHID + col];
      float hv = H[(size_t)(row0 + m) * DHID + col];
      float nn = tanh_fast(np + hn[v] + bhn);
      out[(size_t)(row0 + m) * DHID + col] = (1.0f - z) * nn + z * hv;
    }
  }
}

extern "C" void kernel_launch(void* const* d_in, const int* in_sizes, int n_in,
                              void* d_out, int out_size, void* d_ws, size_t ws_size,
                              hipStream_t stream) {
  const float* X = (const float*)d_in[0];
  const float* S = (const float*)d_in[1];
  const float* H = (const float*)d_in[2];

  // input order: input, seq, hidden, then (W,b) for ir,iz,in_,sr,sz,sn,hr,hz,hn
  unsigned short* Wt = (unsigned short*)d_ws;  // 9 * 16384 bf16 = 288 KB
  const int widx[9] = {3, 5, 7, 9, 11, 13, 15, 17, 19};
  for (int g = 0; g < 9; ++g) {
    transpose_to_bf16<<<MATSZ / 256, 256, 0, stream>>>(
        (const float*)d_in[widx[g]], Wt + (size_t)g * MATSZ);
  }

  const int nblocks = BATCH / 64;  // 16 rows/wave * 4 waves/WG
  gru_fused<<<nblocks, 128, 0, stream>>>(
      X, S, H, Wt,
      (const float*)d_in[4],  (const float*)d_in[6],  (const float*)d_in[8],
      (const float*)d_in[10], (const float*)d_in[12], (const float*)d_in[14],
      (const float*)d_in[16], (const float*)d_in[18], (const float*)d_in[20],
      (float*)d_out);
}
